// SweepGater_77163382440021
// MI455X (gfx1250) — matrix operand order repaired and checked by reading, not compiled, not run on hardware
//
#include <hip/hip_runtime.h>
#include <math.h>

typedef _Float16 v16h  __attribute__((ext_vector_type(16)));
typedef _Float16 v8h   __attribute__((ext_vector_type(8)));
typedef _Float16 f16x2 __attribute__((ext_vector_type(2)));
typedef float    v8f   __attribute__((ext_vector_type(8)));

#define HW        65536   // 256*256
#define CCH       64
#define SN        3
#define NBATCH    4
#define PIXB      128     // pixels per block
#define LDSS      72      // padded row stride in f16 (144B, 16B aligned, bank-friendly)

__device__ __forceinline__ float fast_rcp(float x)  { return __builtin_amdgcn_rcpf(x); }
__device__ __forceinline__ float fast_sqrt(float x) { return __builtin_amdgcn_sqrtf(x); }

// ---------------- kernel 0: zero global accumulators ----------------
__global__ __launch_bounds__(64) void sg_init(float* __restrict__ acc) {
    if (threadIdx.x < 16) acc[threadIdx.x] = 0.0f;
}

// ---------------- kernel 1: WMMA GEMM + sumsq + per-sweep partial sums ----------------
__global__ __launch_bounds__(256) void sg_gemm(
    const float* __restrict__ S, const float* __restrict__ T,
    const float* __restrict__ mask, const float* __restrict__ Wad,
    const float* __restrict__ sigma,
    float* __restrict__ sumsq_out, float* __restrict__ acc_g)
{
    __shared__ __align__(16) _Float16 diff[PIXB][LDSS];  // [pixel][channel]
    __shared__ __align__(16) _Float16 wmat[CCH][LDSS];   // [o][c], pre-scaled by 1/sigma[o]
    __shared__ float redN, redD;

    const int tid  = threadIdx.x;
    const int pt   = blockIdx.x;            // 0..511 pixel tile
    const int bs   = blockIdx.y;            // 0..11
    const int b    = bs / SN, s = bs % SN;
    const int pix0 = pt * PIXB;

    const float* Tb = T + (size_t)(b * SN + s) * CCH * HW;
    const float* Sb = S + (size_t)b * CCH * HW;
    const float* Wb = Wad + s * CCH * CCH;
    const float* sigb = sigma + s * CCH;

    if (tid == 0) { redN = 0.0f; redD = 0.0f; }

    // ---- stage diff = T - S as f16, transposed to [pixel][channel] ----
    #pragma unroll
    for (int i = 0; i < 8; ++i) {
        int cell = i * 256 + tid;
        int pp = (cell & 63) * 2;           // pixel pair
        int cc = (cell >> 6) * 2;           // channel pair
        float2 t0 = *(const float2*)(Tb + cc * HW + pix0 + pp);
        float2 t1 = *(const float2*)(Tb + (cc + 1) * HW + pix0 + pp);
        float2 s0 = *(const float2*)(Sb + cc * HW + pix0 + pp);
        float2 s1 = *(const float2*)(Sb + (cc + 1) * HW + pix0 + pp);
        f16x2 d0 = { (_Float16)(t0.x - s0.x), (_Float16)(t1.x - s1.x) };
        f16x2 d1 = { (_Float16)(t0.y - s0.y), (_Float16)(t1.y - s1.y) };
        *(f16x2*)&diff[pp][cc]     = d0;
        *(f16x2*)&diff[pp + 1][cc] = d1;
    }
    // ---- stage W_ad[s] row-scaled by 1/sigma[o] as f16 [o][c] ----
    // delta = (W . d)/sigma  ==  ((W/sigma_row) . d); fold the divide here, once.
    #pragma unroll
    for (int i = 0; i < 8; ++i) {
        int cell = i * 256 + tid;           // (o, c/2)
        int o  = cell >> 5;
        int c2 = (cell & 31) * 2;
        float isg = fast_rcp(sigb[o]);      // single v_rcp_f32
        float2 wv2 = *(const float2*)(Wb + o * CCH + c2);
        f16x2 wp = { (_Float16)(wv2.x * isg), (_Float16)(wv2.y * isg) };
        *(f16x2*)&wmat[o][c2] = wp;
    }
    __syncthreads();

    const int wvid = tid >> 5;              // wave 0..7
    const int lane = tid & 31;
    const int col  = lane & 15;             // N / C-D column
    const int hi   = lane >> 4;
    const int p    = wvid * 16 + col;       // local pixel this lane's column maps to

    // ---- B operands: v16h[0..7] = K 8*hi+0..7 ; v16h[8..15] = K 16+8*hi+0..7 (matches A) ----
    v8h b0a = *(const v8h*)&diff[p][hi * 8];
    v8h b0b = *(const v8h*)&diff[p][16 + hi * 8];
    v8h b1a = *(const v8h*)&diff[p][32 + hi * 8];
    v8h b1b = *(const v8h*)&diff[p][48 + hi * 8];
    v16h bk0 = __builtin_shufflevector(b0a, b0b, 0,1,2,3,4,5,6,7,8,9,10,11,12,13,14,15);
    v16h bk1 = __builtin_shufflevector(b1a, b1b, 0,1,2,3,4,5,6,7,8,9,10,11,12,13,14,15);

    float sumsq = 0.0f;
    #pragma unroll
    for (int ot = 0; ot < 4; ++ot) {
        const int r = ot * 16 + col;        // A row
        // A layout: v16h[0..7] = K 8*hi+0..7 ; v16h[8..15] = K 16+8*hi+0..7 (per 32-K step)
        v8h a0lo = *(const v8h*)&wmat[r][hi * 8];
        v8h a0hi = *(const v8h*)&wmat[r][16 + hi * 8];
        v8h a1lo = *(const v8h*)&wmat[r][32 + hi * 8];
        v8h a1hi = *(const v8h*)&wmat[r][48 + hi * 8];
        v16h a0 = __builtin_shufflevector(a0lo, a0hi, 0,1,2,3,4,5,6,7,8,9,10,11,12,13,14,15);
        v16h a1 = __builtin_shufflevector(a1lo, a1hi, 0,1,2,3,4,5,6,7,8,9,10,11,12,13,14,15);

        v8f acc = {0.f,0.f,0.f,0.f,0.f,0.f,0.f,0.f};
        acc = __builtin_amdgcn_wmma_f32_16x16x32_f16(false, a0, false, bk0,
                                                     (short)0, acc, false, false);
        acc = __builtin_amdgcn_wmma_f32_16x16x32_f16(false, a1, false, bk1,
                                                     (short)0, acc, false, false);

        #pragma unroll
        for (int v = 0; v < 8; ++v)
            sumsq = fmaf(acc[v], acc[v], sumsq);
    }
    // combine the two half-wave row groups (same pixel column)
    sumsq += __shfl_xor(sumsq, 16, 32);

    const int gpix = pix0 + p;
    const float mval = mask[(size_t)bs * HW + gpix];
    const float w   = fmaxf(mval, 0.0f);
    // l2 = w*sumsq/max(w,1e-6): exactly sumsq when w>=eps, else clipped-denominator branch
    const float l2  = (w >= 1e-6f) ? sumsq : w * sumsq * 1e6f;
    if (hi == 0) sumsq_out[(size_t)bs * HW + gpix] = sumsq;

    // per-sweep global accumulation of l2*w and w
    float cn = (hi == 0) ? l2 * w : 0.0f;
    float cd = (hi == 0) ? w : 0.0f;
    #pragma unroll
    for (int off = 16; off > 0; off >>= 1) {
        cn += __shfl_xor(cn, off, 32);
        cd += __shfl_xor(cd, off, 32);
    }
    if (lane == 0) { atomicAdd(&redN, cn); atomicAdd(&redD, cd); }
    __syncthreads();
    if (tid == 0) {
        atomicAdd(&acc_g[2 * s],     redN);
        atomicAdd(&acc_g[2 * s + 1], redD);
    }
}

// ---------------- kernel 2: per-pixel masked softmax over sweeps ----------------
__global__ __launch_bounds__(256) void sg_softmax(
    const float* __restrict__ mask, const float* __restrict__ sumsq_ws,
    const float* __restrict__ acc_g, float* __restrict__ out)
{
    const int t = blockIdx.x * 256 + threadIdx.x;    // 0..65535 (4 pixels each)
    const int b = t >> 14;
    const int base = (t & 16383) * 4;

    float l2m[SN];
    #pragma unroll
    for (int s = 0; s < SN; ++s)
        l2m[s] = acc_g[2 * s] * fast_rcp(fmaxf(acc_g[2 * s + 1], 1e-6f));

    float4 ss[SN], mm[SN];
    #pragma unroll
    for (int s = 0; s < SN; ++s) {
        size_t idx = (size_t)(b * SN + s) * HW + base;
        ss[s] = *(const float4*)(sumsq_ws + idx);
        mm[s] = *(const float4*)(mask + idx);
    }

    float av[SN][4];
    #pragma unroll
    for (int j = 0; j < 4; ++j) {
        float sc[SN], wv[SN];
        float mx = -INFINITY;
        #pragma unroll
        for (int s = 0; s < SN; ++s) {
            float w  = fmaxf(((const float*)&mm[s])[j], 0.0f);
            float sq = ((const float*)&ss[s])[j];
            float l2 = (w >= 1e-6f) ? sq : w * sq * 1e6f;
            float rdn = fast_rcp(fast_sqrt(sq) + 1e-4f);
            float score = (w <= 0.0f) ? -INFINITY
                                      : (l2m[s] - l2) * rdn * (1.0f / 0.7f);
            wv[s] = w; sc[s] = score;
            mx = fmaxf(mx, score);
        }
        float z = 0.0f;
        #pragma unroll
        for (int s = 0; s < SN; ++s) {
            float e = (sc[s] == -INFINITY) ? 0.0f : __expf(sc[s] - mx);
            if (wv[s] <= 0.0f) e = 0.0f;
            av[s][j] = e; z += e;
        }
        float rz = fast_rcp(fmaxf(z, 1e-6f));
        #pragma unroll
        for (int s = 0; s < SN; ++s) av[s][j] *= rz;
    }
    #pragma unroll
    for (int s = 0; s < SN; ++s) {
        size_t idx = (size_t)(b * SN + s) * HW + base;
        *(float4*)(out + idx) = make_float4(av[s][0], av[s][1], av[s][2], av[s][3]);
    }
}

extern "C" void kernel_launch(void* const* d_in, const int* in_sizes, int n_in,
                              void* d_out, int out_size, void* d_ws, size_t ws_size,
                              hipStream_t stream) {
    const float* S     = (const float*)d_in[0];
    const float* T     = (const float*)d_in[1];
    const float* mask  = (const float*)d_in[2];
    const float* Wad   = (const float*)d_in[3];
    // d_in[4] = b_ad   (cancels in delta)
    // d_in[5] = ema_mu (cancels in delta)
    const float* sigma = (const float*)d_in[6];

    float* acc_g    = (float*)d_ws;          // 16 floats (6 used)
    float* sumsq_ws = (float*)d_ws + 16;     // B*SN*HW floats = 3 MB

    hipLaunchKernelGGL(sg_init, dim3(1), dim3(64), 0, stream, acc_g);
    hipLaunchKernelGGL(sg_gemm, dim3(HW / PIXB, NBATCH * SN), dim3(256), 0, stream,
                       S, T, mask, Wad, sigma, sumsq_ws, acc_g);
    hipLaunchKernelGGL(sg_softmax, dim3((NBATCH * HW / 4) / 256), dim3(256), 0, stream,
                       mask, sumsq_ws, acc_g, (float*)d_out);
}
